// MAGIKNet_26963804684931
// MI455X (gfx1250) — matrix operand order repaired and checked by reading, not compiled, not run on hardware
//
#include <hip/hip_runtime.h>
#include <hip/hip_bf16.h>
#include <hip/hip_fp16.h>

// ---------------------------------------------------------------------------
// MAGIK-style GNN forward on gfx1250 (MI455X), wave32 + v_wmma_f32_16x16x32_f16.
//
// Input flattening assumption (jax pytree: dict keys sorted, tuples in order):
//  0 node_features (N,5) f32   1 edge_features (E,4) f32   2 edge_index (2,E) i32
//  3..12  cls:  bn1.g, bn1.b, bn2.g, bn2.b, l1.W(132x64), l1.b, l2.W(64x32), l2.b, l3.W(32x1), l3.b
//  13+8l.. convs[l]: bn1.g, bn1.b, bn2.g, bn2.b, l1.W(128x64), l1.b, l2.W(64x64), l2.b
//  37..44 node_enc: bn1.g, bn1.b, bn2.g, bn2.b, l1.W(5x64), l1.b, l2.W(64x64), l2.b
// ---------------------------------------------------------------------------

typedef __attribute__((ext_vector_type(16))) _Float16 v16h;
typedef __attribute__((ext_vector_type(8)))  _Float16 v8h;
typedef __attribute__((ext_vector_type(8)))  float    v8f;

#define EPSBN 1e-5f

__device__ __forceinline__ v8f wmma16(v16h a, v16h b, v8f c) {
  // D = A(16x32 f16) x B(32x16 f16) + C(16x16 f32)
  return __builtin_amdgcn_wmma_f32_16x16x32_f16(false, a, false, b, (short)0, c,
                                                false, false);
}

__device__ __forceinline__ v16h pack16(v8h lo, v8h hi) {
  v16h a;
#pragma unroll
  for (int i = 0; i < 8; ++i) { a[i] = lo[i]; a[8 + i] = hi[i]; }
  return a;
}

// B fragment for 16-bit 32x16: lane (l&15)=col; K groups of 8 at step + (l>>4)*8
// and step + 16 + (l>>4)*8 (kbase = step + (l>>4)*16); Wt[n*Kpad + k] = W[k*N + n].
__device__ __forceinline__ v16h load_b_frag(const _Float16* __restrict__ Wt,
                                            int Kpad, int col, int kbase) {
  const _Float16* p = Wt + (size_t)col * Kpad + (kbase & ~31) + ((kbase & 31) >> 1);
  return pack16(*(const v8h*)p, *(const v8h*)(p + 16));
}

// ------------------------------ small kernels ------------------------------

__global__ void k_zero(float* __restrict__ p, long long n) {
  long long i = (long long)blockIdx.x * blockDim.x + threadIdx.x;
  if (i < n) p[i] = 0.f;
}

// dst[n*Kpad + k] = (k < K) ? (f16)W[k*N + n] : 0
__global__ void k_convt(const float* __restrict__ W, _Float16* __restrict__ dst,
                        int K, int Nn, int Kpad) {
  int i = blockIdx.x * blockDim.x + threadIdx.x;
  if (i < Nn * Kpad) {
    int n = i / Kpad, k = i - n * Kpad;
    dst[i] = (k < K) ? (_Float16)W[k * Nn + n] : (_Float16)0.f;
  }
}

// node encoder linear1: (N,5) @ (5,64) + b  -> out, accumulate column stats
__global__ __launch_bounds__(256) void k_lin5(
    const float* __restrict__ nf, const float* __restrict__ W,
    const float* __restrict__ b, float* __restrict__ out,
    float* __restrict__ statsA, int N) {
  __shared__ float s_sum[64], s_sq[64];
  int tid = threadIdx.x;
  if (tid < 64) { s_sum[tid] = 0.f; s_sq[tid] = 0.f; }
  __syncthreads();
  long long i = (long long)blockIdx.x * blockDim.x + tid;
  if (i < (long long)N * 64) {
    int c = (int)(i & 63);
    long long row = i >> 6;
    float acc = b[c];
#pragma unroll
    for (int k = 0; k < 5; ++k) acc += nf[row * 5 + k] * W[k * 64 + c];
    out[i] = acc;
    atomicAdd(&s_sum[c], acc);
    atomicAdd(&s_sq[c], acc * acc);
  }
  __syncthreads();
  if (tid < 64) {
    unsafeAtomicAdd(&statsA[tid], s_sum[tid]);
    unsafeAtomicAdd(&statsA[64 + tid], s_sq[tid]);
  }
}

// statsA(sum,sumsq) -> statsB(scale,shift); reset statsA for next use
__global__ void k_finalize_bn(float* __restrict__ statsA, float* __restrict__ statsB,
                              const float* __restrict__ g, const float* __restrict__ beta,
                              float cnt, int C) {
  int c = threadIdx.x;
  if (c < C) {
    float mean = statsA[c] / cnt;
    float var  = statsA[64 + c] / cnt - mean * mean;  // biased, matches ref
    float scale = g[c] * rsqrtf(var + EPSBN);
    statsB[c]      = scale;
    statsB[64 + c] = beta[c] - mean * scale;
    statsA[c] = 0.f;
    statsA[64 + c] = 0.f;
  }
}

// x = relu(x*scale+shift); also emit f16 copy
__global__ void k_bn_relu_node(float* __restrict__ x, _Float16* __restrict__ xh,
                               const float* __restrict__ sB, long long n) {
  long long i = (long long)blockIdx.x * blockDim.x + threadIdx.x;
  if (i < n) {
    int c = (int)(i & 63);
    float v = fmaxf(x[i] * sB[c] + sB[64 + c], 0.f);
    x[i] = v;
    xh[i] = (_Float16)v;
  }
}

// x = relu(x + upd); emit f16 copy
__global__ void k_residual(float* __restrict__ x, const float* __restrict__ upd,
                           _Float16* __restrict__ xh, long long n) {
  long long i = (long long)blockIdx.x * blockDim.x + threadIdx.x;
  if (i < n) {
    float v = fmaxf(x[i] + upd[i], 0.f);
    x[i] = v;
    xh[i] = (_Float16)v;
  }
}

// scatter-add relu(bn(yh)) into upd[dst]
__global__ void k_scatter(const _Float16* __restrict__ yh, const float* __restrict__ sB,
                          const int* __restrict__ dstI, float* __restrict__ upd,
                          long long n) {
  long long i = (long long)blockIdx.x * blockDim.x + threadIdx.x;
  if (i < n) {
    int c = (int)(i & 63);
    long long e = i >> 6;
    float v = fmaxf((float)yh[i] * sB[c] + sB[64 + c], 0.f);
    unsafeAtomicAdd(&upd[(size_t)dstI[e] * 64 + c], v);
  }
}

// final classifier linear: logit = relu(bn(h2)) . W3 + b3
__global__ void k_cls3(const _Float16* __restrict__ yh, const float* __restrict__ sB,
                       const float* __restrict__ W3, const float* __restrict__ b3,
                       float* __restrict__ out, int E) {
  int e = blockIdx.x * blockDim.x + threadIdx.x;
  if (e < E) {
    const _Float16* row = yh + (size_t)e * 64;
    float acc = b3[0];
#pragma unroll
    for (int c = 0; c < 32; ++c) {
      float h = fmaxf((float)row[c] * sB[c] + sB[64 + c], 0.f);
      acc += h * W3[c];
    }
    out[e] = acc;
  }
}

// ------------------------------ WMMA kernels -------------------------------
// One wave computes a 16(M) x 64(N) stripe: 4 accumulators, K-loop of 32.
// A layout (16-bit 16x32): lane row = lane&15; K groups of 8 at
//   ks*32 + (lane>>4)*8  and  ks*32 + 16 + (lane>>4)*8.
// C/D layout: VGPR r, lane -> (row = r + (lane>>4)*8, col = lane&15).

// EdgeConv linear1: A = [xh[src] || xh[dst]] (K=128) @ W1^T
__global__ __launch_bounds__(256) void k_edge_mlp1(
    const _Float16* __restrict__ xh, const int* __restrict__ srcI,
    const int* __restrict__ dstI, const _Float16* __restrict__ Wt,
    const float* __restrict__ bias, _Float16* __restrict__ yh,
    float* __restrict__ statsA, int E) {
  __shared__ float s_sum[64], s_sq[64];
  int tid = threadIdx.x;
  if (tid < 64) { s_sum[tid] = 0.f; s_sq[tid] = 0.f; }
  __syncthreads();
  int wave = (int)(((long long)blockIdx.x * blockDim.x + tid) >> 5);
  int lane = tid & 31;
  int m0 = wave * 16;
  if (m0 < E) {
    int lr = lane & 15, hi = lane >> 4;
    int e = m0 + lr;
    const _Float16* ps = xh + (size_t)srcI[e] * 64;
    const _Float16* pd = xh + (size_t)dstI[e] * 64;
    v8f acc[4];
#pragma unroll
    for (int t = 0; t < 4; ++t)
#pragma unroll
      for (int r = 0; r < 8; ++r) acc[t][r] = 0.f;
#pragma unroll
    for (int ks = 0; ks < 4; ++ks) {
      int g0 = ks * 32 + hi * 8;
      int g1 = g0 + 16;
      v8h lo = (g0 < 64) ? *(const v8h*)(ps + g0) : *(const v8h*)(pd + (g0 - 64));
      v8h hh = (g1 < 64) ? *(const v8h*)(ps + g1) : *(const v8h*)(pd + (g1 - 64));
      v16h a = pack16(lo, hh);
      int kbase = ks * 32 + hi * 16;
#pragma unroll
      for (int t = 0; t < 4; ++t)
        acc[t] = wmma16(a, load_b_frag(Wt, 128, lr + 16 * t, kbase), acc[t]);
    }
#pragma unroll
    for (int t = 0; t < 4; ++t) {
      int col = lr + 16 * t;
      float bb = bias[col];
      float lsum = 0.f, lsq = 0.f;
#pragma unroll
      for (int r = 0; r < 8; ++r) {
        float v = acc[t][r] + bb;
        yh[(size_t)(m0 + r + hi * 8) * 64 + col] = (_Float16)v;
        lsum += v;
        lsq += v * v;
      }
      atomicAdd(&s_sum[col], lsum);
      atomicAdd(&s_sq[col], lsq);
    }
  }
  __syncthreads();
  if (tid < 64) {
    unsafeAtomicAdd(&statsA[tid], s_sum[tid]);
    unsafeAtomicAdd(&statsA[64 + tid], s_sq[tid]);
  }
}

// A = relu(bn(io f16, K=64)) @ W^T -> io (in-place safe: wave reads full tile first)
__global__ __launch_bounds__(256) void k_edge_mlp2(
    _Float16* __restrict__ io, const float* __restrict__ sB,
    const _Float16* __restrict__ Wt, const float* __restrict__ bias,
    float* __restrict__ statsA, int rows) {
  __shared__ float s_sum[64], s_sq[64];
  int tid = threadIdx.x;
  if (tid < 64) { s_sum[tid] = 0.f; s_sq[tid] = 0.f; }
  __syncthreads();
  int wave = (int)(((long long)blockIdx.x * blockDim.x + tid) >> 5);
  int lane = tid & 31;
  int m0 = wave * 16;
  if (m0 < rows) {
    int lr = lane & 15, hi = lane >> 4;
    const _Float16* arow = io + (size_t)(m0 + lr) * 64;
    v8f acc[4];
#pragma unroll
    for (int t = 0; t < 4; ++t)
#pragma unroll
      for (int r = 0; r < 8; ++r) acc[t][r] = 0.f;
#pragma unroll
    for (int ks = 0; ks < 2; ++ks) {
      int g0 = ks * 32 + hi * 8;
      int g1 = g0 + 16;
      v8h r0 = *(const v8h*)(arow + g0);
      v8h r1 = *(const v8h*)(arow + g1);
      v8h lo, hh;
#pragma unroll
      for (int i = 0; i < 8; ++i) {
        lo[i] = (_Float16)fmaxf((float)r0[i] * sB[g0 + i] + sB[64 + g0 + i], 0.f);
        hh[i] = (_Float16)fmaxf((float)r1[i] * sB[g1 + i] + sB[64 + g1 + i], 0.f);
      }
      v16h a = pack16(lo, hh);
      int kbase = ks * 32 + hi * 16;
#pragma unroll
      for (int t = 0; t < 4; ++t)
        acc[t] = wmma16(a, load_b_frag(Wt, 64, lr + 16 * t, kbase), acc[t]);
    }
#pragma unroll
    for (int t = 0; t < 4; ++t) {
      int col = lr + 16 * t;
      float bb = bias[col];
      float lsum = 0.f, lsq = 0.f;
#pragma unroll
      for (int r = 0; r < 8; ++r) {
        float v = acc[t][r] + bb;
        io[(size_t)(m0 + r + hi * 8) * 64 + col] = (_Float16)v;
        lsum += v;
        lsq += v * v;
      }
      atomicAdd(&s_sum[col], lsum);
      atomicAdd(&s_sq[col], lsq);
    }
  }
  __syncthreads();
  if (tid < 64) {
    unsafeAtomicAdd(&statsA[tid], s_sum[tid]);
    unsafeAtomicAdd(&statsA[64 + tid], s_sq[tid]);
  }
}

// node encoder linear2: A = relu(bn(f32 buf, K=64)) @ W^T -> f32 out
__global__ __launch_bounds__(256) void k_ne2(
    const float* __restrict__ A, const float* __restrict__ sB,
    const _Float16* __restrict__ Wt, const float* __restrict__ bias,
    float* __restrict__ out, float* __restrict__ statsA, int rows) {
  __shared__ float s_sum[64], s_sq[64];
  int tid = threadIdx.x;
  if (tid < 64) { s_sum[tid] = 0.f; s_sq[tid] = 0.f; }
  __syncthreads();
  int wave = (int)(((long long)blockIdx.x * blockDim.x + tid) >> 5);
  int lane = tid & 31;
  int m0 = wave * 16;
  if (m0 < rows) {
    int lr = lane & 15, hi = lane >> 4;
    const float* arow = A + (size_t)(m0 + lr) * 64;
    v8f acc[4];
#pragma unroll
    for (int t = 0; t < 4; ++t)
#pragma unroll
      for (int r = 0; r < 8; ++r) acc[t][r] = 0.f;
#pragma unroll
    for (int ks = 0; ks < 2; ++ks) {
      int g0 = ks * 32 + hi * 8;
      int g1 = g0 + 16;
      v8h lo, hh;
#pragma unroll
      for (int i = 0; i < 8; ++i) {
        lo[i] = (_Float16)fmaxf(arow[g0 + i] * sB[g0 + i] + sB[64 + g0 + i], 0.f);
        hh[i] = (_Float16)fmaxf(arow[g1 + i] * sB[g1 + i] + sB[64 + g1 + i], 0.f);
      }
      v16h a = pack16(lo, hh);
      int kbase = ks * 32 + hi * 16;
#pragma unroll
      for (int t = 0; t < 4; ++t)
        acc[t] = wmma16(a, load_b_frag(Wt, 64, lr + 16 * t, kbase), acc[t]);
    }
#pragma unroll
    for (int t = 0; t < 4; ++t) {
      int col = lr + 16 * t;
      float bb = bias[col];
      float lsum = 0.f, lsq = 0.f;
#pragma unroll
      for (int r = 0; r < 8; ++r) {
        float v = acc[t][r] + bb;
        out[(size_t)(m0 + r + hi * 8) * 64 + col] = v;
        lsum += v;
        lsq += v * v;
      }
      atomicAdd(&s_sum[col], lsum);
      atomicAdd(&s_sq[col], lsq);
    }
  }
  __syncthreads();
  if (tid < 64) {
    unsafeAtomicAdd(&statsA[tid], s_sum[tid]);
    unsafeAtomicAdd(&statsA[64 + tid], s_sq[tid]);
  }
}

// classifier linear1: A = [xh[src] || xh[dst] || ef || pad] (K=160) @ W^T
__device__ __forceinline__ v8h load_cls_grp(const _Float16* __restrict__ ps,
                                            const _Float16* __restrict__ pd,
                                            const float* __restrict__ ef, int g) {
  if (g < 64) return *(const v8h*)(ps + g);
  if (g < 128) return *(const v8h*)(pd + (g - 64));
  v8h r;
#pragma unroll
  for (int i = 0; i < 8; ++i) r[i] = (_Float16)0.f;
  if (g == 128) {
#pragma unroll
    for (int i = 0; i < 4; ++i) r[i] = (_Float16)ef[i];
  }
  return r;
}

__global__ __launch_bounds__(256) void k_cls1(
    const _Float16* __restrict__ xh, const int* __restrict__ srcI,
    const int* __restrict__ dstI, const float* __restrict__ ef,
    const _Float16* __restrict__ Wt, const float* __restrict__ bias,
    _Float16* __restrict__ yh, float* __restrict__ statsA, int E) {
  __shared__ float s_sum[64], s_sq[64];
  int tid = threadIdx.x;
  if (tid < 64) { s_sum[tid] = 0.f; s_sq[tid] = 0.f; }
  __syncthreads();
  int wave = (int)(((long long)blockIdx.x * blockDim.x + tid) >> 5);
  int lane = tid & 31;
  int m0 = wave * 16;
  if (m0 < E) {
    int lr = lane & 15, hi = lane >> 4;
    int e = m0 + lr;
    const _Float16* ps = xh + (size_t)srcI[e] * 64;
    const _Float16* pd = xh + (size_t)dstI[e] * 64;
    const float* pe = ef + (size_t)e * 4;
    v8f acc[4];
#pragma unroll
    for (int t = 0; t < 4; ++t)
#pragma unroll
      for (int r = 0; r < 8; ++r) acc[t][r] = 0.f;
#pragma unroll
    for (int ks = 0; ks < 5; ++ks) {
      int g0 = ks * 32 + hi * 8;
      int g1 = g0 + 16;
      v16h a = pack16(load_cls_grp(ps, pd, pe, g0), load_cls_grp(ps, pd, pe, g1));
      int kbase = ks * 32 + hi * 16;
#pragma unroll
      for (int t = 0; t < 4; ++t)
        acc[t] = wmma16(a, load_b_frag(Wt, 160, lr + 16 * t, kbase), acc[t]);
    }
#pragma unroll
    for (int t = 0; t < 4; ++t) {
      int col = lr + 16 * t;
      float bb = bias[col];
      float lsum = 0.f, lsq = 0.f;
#pragma unroll
      for (int r = 0; r < 8; ++r) {
        float v = acc[t][r] + bb;
        yh[(size_t)(m0 + r + hi * 8) * 64 + col] = (_Float16)v;
        lsum += v;
        lsq += v * v;
      }
      atomicAdd(&s_sum[col], lsum);
      atomicAdd(&s_sq[col], lsq);
    }
  }
  __syncthreads();
  if (tid < 64) {
    unsafeAtomicAdd(&statsA[tid], s_sum[tid]);
    unsafeAtomicAdd(&statsA[64 + tid], s_sq[tid]);
  }
}

// classifier linear2: A = relu(bn(yh, K=64)) @ W^T (64->32), out into cols 0..31
__global__ __launch_bounds__(256) void k_cls2(
    _Float16* __restrict__ io, const float* __restrict__ sB,
    const _Float16* __restrict__ Wt, const float* __restrict__ bias,
    float* __restrict__ statsA, int rows) {
  __shared__ float s_sum[32], s_sq[32];
  int tid = threadIdx.x;
  if (tid < 32) { s_sum[tid] = 0.f; s_sq[tid] = 0.f; }
  __syncthreads();
  int wave = (int)(((long long)blockIdx.x * blockDim.x + tid) >> 5);
  int lane = tid & 31;
  int m0 = wave * 16;
  if (m0 < rows) {
    int lr = lane & 15, hi = lane >> 4;
    const _Float16* arow = io + (size_t)(m0 + lr) * 64;
    v8f acc[2];
#pragma unroll
    for (int t = 0; t < 2; ++t)
#pragma unroll
      for (int r = 0; r < 8; ++r) acc[t][r] = 0.f;
#pragma unroll
    for (int ks = 0; ks < 2; ++ks) {
      int g0 = ks * 32 + hi * 8;
      int g1 = g0 + 16;
      v8h r0 = *(const v8h*)(arow + g0);
      v8h r1 = *(const v8h*)(arow + g1);
      v8h lo, hh;
#pragma unroll
      for (int i = 0; i < 8; ++i) {
        lo[i] = (_Float16)fmaxf((float)r0[i] * sB[g0 + i] + sB[64 + g0 + i], 0.f);
        hh[i] = (_Float16)fmaxf((float)r1[i] * sB[g1 + i] + sB[64 + g1 + i], 0.f);
      }
      v16h a = pack16(lo, hh);
      int kbase = ks * 32 + hi * 16;
#pragma unroll
      for (int t = 0; t < 2; ++t)
        acc[t] = wmma16(a, load_b_frag(Wt, 64, lr + 16 * t, kbase), acc[t]);
    }
#pragma unroll
    for (int t = 0; t < 2; ++t) {
      int col = lr + 16 * t;  // 0..31
      float bb = bias[col];
      float lsum = 0.f, lsq = 0.f;
#pragma unroll
      for (int r = 0; r < 8; ++r) {
        float v = acc[t][r] + bb;
        io[(size_t)(m0 + r + hi * 8) * 64 + col] = (_Float16)v;
        lsum += v;
        lsq += v * v;
      }
      atomicAdd(&s_sum[col], lsum);
      atomicAdd(&s_sq[col], lsq);
    }
  }
  __syncthreads();
  if (tid < 32) {
    unsafeAtomicAdd(&statsA[tid], s_sum[tid]);
    unsafeAtomicAdd(&statsA[64 + tid], s_sq[tid]);
  }
}

// ------------------------------- host side ---------------------------------

extern "C" void kernel_launch(void* const* d_in, const int* in_sizes, int n_in,
                              void* d_out, int out_size, void* d_ws, size_t ws_size,
                              hipStream_t stream) {
  (void)out_size;
  (void)ws_size;
  if (n_in < 45) return;

  const float* node_f = (const float*)d_in[0];
  const float* edge_f = (const float*)d_in[1];
  const int* eidx = (const int*)d_in[2];
  const int N = in_sizes[0] / 5;
  const int E = in_sizes[1] / 4;
  const int* srcI = eidx;
  const int* dstI = eidx + E;

  const float* cls_bn1_g = (const float*)d_in[3];
  const float* cls_bn1_b = (const float*)d_in[4];
  const float* cls_bn2_g = (const float*)d_in[5];
  const float* cls_bn2_b = (const float*)d_in[6];
  const float* cls_l1_W = (const float*)d_in[7];
  const float* cls_l1_b = (const float*)d_in[8];
  const float* cls_l2_W = (const float*)d_in[9];
  const float* cls_l2_b = (const float*)d_in[10];
  const float* cls_l3_W = (const float*)d_in[11];
  const float* cls_l3_b = (const float*)d_in[12];

  const float* ne_bn1_g = (const float*)d_in[37];
  const float* ne_bn1_b = (const float*)d_in[38];
  const float* ne_bn2_g = (const float*)d_in[39];
  const float* ne_bn2_b = (const float*)d_in[40];
  const float* ne_l1_W = (const float*)d_in[41];
  const float* ne_l1_b = (const float*)d_in[42];
  const float* ne_l2_W = (const float*)d_in[43];
  const float* ne_l2_b = (const float*)d_in[44];

  // workspace carve-out (256B aligned slices)
  char* wsb = (char*)d_ws;
  size_t off = 0;
  auto alloc = [&](size_t bytes) -> char* {
    char* p = wsb + off;
    off += (bytes + 255) & ~(size_t)255;
    return p;
  };
  _Float16* xh = (_Float16*)alloc((size_t)N * 64 * 2);
  float* x = (float*)alloc((size_t)N * 64 * 4);
  float* nbuf = (float*)alloc((size_t)N * 64 * 4);  // node-enc tmp, then upd
  _Float16* yh = (_Float16*)alloc((size_t)E * 64 * 2);
  _Float16* w_ne2 = (_Float16*)alloc(64 * 64 * 2);
  _Float16* w_c1[3];
  _Float16* w_c2[3];
  for (int l = 0; l < 3; ++l) {
    w_c1[l] = (_Float16*)alloc(64 * 128 * 2);
    w_c2[l] = (_Float16*)alloc(64 * 64 * 2);
  }
  _Float16* w_cls1 = (_Float16*)alloc(64 * 160 * 2);
  _Float16* w_cls2 = (_Float16*)alloc(32 * 64 * 2);
  float* statsA = (float*)alloc(128 * 4);
  float* statsB = (float*)alloc(128 * 4);

  auto cdiv = [](long long a, long long b) { return (int)((a + b - 1) / b); };

  // weight conversion (f32 -> f16, transposed to [N][Kpad])
  k_convt<<<cdiv(64 * 64, 256), 256, 0, stream>>>(ne_l2_W, w_ne2, 64, 64, 64);
  for (int l = 0; l < 3; ++l) {
    const float* l1W = (const float*)d_in[13 + 8 * l + 4];
    const float* l2W = (const float*)d_in[13 + 8 * l + 6];
    k_convt<<<cdiv(64 * 128, 256), 256, 0, stream>>>(l1W, w_c1[l], 128, 64, 128);
    k_convt<<<cdiv(64 * 64, 256), 256, 0, stream>>>(l2W, w_c2[l], 64, 64, 64);
  }
  k_convt<<<cdiv(64 * 160, 256), 256, 0, stream>>>(cls_l1_W, w_cls1, 132, 64, 160);
  k_convt<<<cdiv(32 * 64, 256), 256, 0, stream>>>(cls_l2_W, w_cls2, 64, 32, 64);
  k_zero<<<1, 128, 0, stream>>>(statsA, 128);

  const int nblocks = cdiv((N + 15) / 16, 8);  // 8 waves/block
  const int eblocks = cdiv((E + 15) / 16, 8);
  const long long N64 = (long long)N * 64;
  const long long E64 = (long long)E * 64;

  // node encoder
  k_lin5<<<cdiv(N64, 256), 256, 0, stream>>>(node_f, ne_l1_W, ne_l1_b, nbuf, statsA, N);
  k_finalize_bn<<<1, 64, 0, stream>>>(statsA, statsB, ne_bn1_g, ne_bn1_b, (float)N, 64);
  k_ne2<<<nblocks, 256, 0, stream>>>(nbuf, statsB, w_ne2, ne_l2_b, x, statsA, N);
  k_finalize_bn<<<1, 64, 0, stream>>>(statsA, statsB, ne_bn2_g, ne_bn2_b, (float)N, 64);
  k_bn_relu_node<<<cdiv(N64, 256), 256, 0, stream>>>(x, xh, statsB, N64);

  // EdgeConv layers
  for (int l = 0; l < 3; ++l) {
    int base = 13 + 8 * l;
    const float* bn1g = (const float*)d_in[base + 0];
    const float* bn1b = (const float*)d_in[base + 1];
    const float* bn2g = (const float*)d_in[base + 2];
    const float* bn2b = (const float*)d_in[base + 3];
    const float* l1b = (const float*)d_in[base + 5];
    const float* l2b = (const float*)d_in[base + 7];

    k_edge_mlp1<<<eblocks, 256, 0, stream>>>(xh, srcI, dstI, w_c1[l], l1b, yh, statsA, E);
    k_finalize_bn<<<1, 64, 0, stream>>>(statsA, statsB, bn1g, bn1b, (float)E, 64);
    k_edge_mlp2<<<eblocks, 256, 0, stream>>>(yh, statsB, w_c2[l], l2b, statsA, E);
    k_finalize_bn<<<1, 64, 0, stream>>>(statsA, statsB, bn2g, bn2b, (float)E, 64);
    k_zero<<<cdiv(N64, 256), 256, 0, stream>>>(nbuf, N64);
    k_scatter<<<cdiv(E64, 256), 256, 0, stream>>>(yh, statsB, dstI, nbuf, E64);
    k_residual<<<cdiv(N64, 256), 256, 0, stream>>>(x, nbuf, xh, N64);
  }

  // classifier
  k_cls1<<<eblocks, 256, 0, stream>>>(xh, srcI, dstI, edge_f, w_cls1, cls_l1_b, yh, statsA, E);
  k_finalize_bn<<<1, 64, 0, stream>>>(statsA, statsB, cls_bn1_g, cls_bn1_b, (float)E, 64);
  k_cls2<<<eblocks, 256, 0, stream>>>(yh, statsB, w_cls2, cls_l2_b, statsA, E);
  k_finalize_bn<<<1, 64, 0, stream>>>(statsA, statsB, cls_bn2_g, cls_bn2_b, (float)E, 32);
  k_cls3<<<cdiv(E, 256), 256, 0, stream>>>(yh, statsB, cls_l3_W, cls_l3_b, (float*)d_out, E);
}
